// MLP_4913442587016
// MI455X (gfx1250) — matrix operand order repaired and checked by reading, not compiled, not run on hardware
//
#include <hip/hip_runtime.h>

// ---------------------------------------------------------------------------
// MI455X (gfx1250): spmv (COO, coalesced atomics) -> 3x [256x8192]@[8192x8192]
// GEMMs on V_WMMA_F32_16X16X32_BF16 with bf16 hi/lo split (fp32-grade accuracy,
// ~3x bf16 math, still HBM-bound on the 805 MB weight stream).
// Activations transposed [N][B]; W tiles streamed via async global->LDS B128
// copies with double-buffered LDS and one barrier per K step.
// ---------------------------------------------------------------------------

typedef __attribute__((ext_vector_type(16))) __bf16 v16bf;
typedef __attribute__((ext_vector_type(4)))  __bf16 bf16x4;
typedef __attribute__((ext_vector_type(2)))  __bf16 bf16x2;
typedef __attribute__((ext_vector_type(8)))  float  v8f;
typedef int v4i __attribute__((vector_size(16)));   // pointee type for async-LDS builtin

constexpr int N     = 8192;
constexpr int BATCH = 256;
constexpr int NNZ   = 16 * N;
constexpr int BM    = 32;          // W rows per workgroup
constexpr int KB    = 32;          // K per stage
constexpr int WROW  = KB + 12;     // 44 f32 = 176 B row stride (16B aligned, bank-clean)
constexpr int WBUF  = BM * WROW;   // f32 per W buffer
constexpr int LROW  = KB + 8;      // 40 bf16 = 80 B row stride
constexpr int HBUF  = BATCH * LROW;// bf16 per H buffer

union FragBf { v16bf v; bf16x4 c[4]; };

#if defined(__has_builtin)
#if __has_builtin(__builtin_amdgcn_global_load_async_to_lds_b128)
#define HAS_ASYNC_LDS 1
#endif
#endif
#ifndef HAS_ASYNC_LDS
#define HAS_ASYNC_LDS 0
#endif

__device__ __forceinline__ float bf2f(__bf16 h) {
  unsigned short s; __builtin_memcpy(&s, &h, 2);
  return __uint_as_float(((unsigned)s) << 16);
}
__device__ __forceinline__ bf16x2 pk_bf16(float a, float b) {
#if __has_builtin(__builtin_amdgcn_cvt_pk_bf16_f32)
  return __builtin_amdgcn_cvt_pk_bf16_f32(a, b);
#else
  bf16x2 r; r[0] = (__bf16)a; r[1] = (__bf16)b; return r;
#endif
}
// fp32x4 -> bf16 hi + residual lo
__device__ __forceinline__ void cvt_split4(float4 v, bf16x4& hi, bf16x4& lo) {
  bf16x2 h01 = pk_bf16(v.x, v.y), h23 = pk_bf16(v.z, v.w);
  bf16x2 l01 = pk_bf16(v.x - bf2f(h01[0]), v.y - bf2f(h01[1]));
  bf16x2 l23 = pk_bf16(v.z - bf2f(h23[0]), v.w - bf2f(h23[1]));
  hi[0] = h01[0]; hi[1] = h01[1]; hi[2] = h23[0]; hi[3] = h23[1];
  lo[0] = l01[0]; lo[1] = l01[1]; lo[2] = l23[0]; lo[3] = l23[1];
}
__device__ __forceinline__ float selu_f(float x) {
  const float S = 1.0507009873554805f, A = 1.6732632423543772f;
  return x > 0.f ? S * x : S * A * (__expf(x) - 1.f);
}
__device__ __forceinline__ void async_wait() {
#if HAS_ASYNC_LDS
#if __has_builtin(__builtin_amdgcn_s_wait_asynccnt)
  __builtin_amdgcn_s_wait_asynccnt(0);
#else
  asm volatile("s_wait_asynccnt 0x0" ::: "memory");
#endif
#endif
}

// ---------------- x [B][N] -> x_t [N][B] ------------------------------------
__global__ __launch_bounds__(256) void k_transpose(const float* __restrict__ x,
                                                   float* __restrict__ xt) {
  __shared__ float tile[32][33];
  const int bx = blockIdx.x, by = blockIdx.y;
  const int tx = threadIdx.x & 31, ty = threadIdx.x >> 5;
#pragma unroll
  for (int i = 0; i < 32; i += 8)
    tile[ty + i][tx] = x[(size_t)(by * 32 + ty + i) * N + bx * 32 + tx];
  __syncthreads();
#pragma unroll
  for (int i = 0; i < 32; i += 8)
    xt[(size_t)(bx * 32 + ty + i) * BATCH + by * 32 + tx] = tile[tx][ty + i];
}

// ---------------- COO scatter: one wave per nnz -----------------------------
__global__ __launch_bounds__(256) void k_spmv(const int* __restrict__ rows,
                                              const int* __restrict__ cols,
                                              const float* __restrict__ vals,
                                              const float* __restrict__ xt,
                                              float* __restrict__ yt) {
  const int wave = threadIdx.x >> 5, lane = threadIdx.x & 31;
  const int k = blockIdx.x * 8 + wave;
  const int r = rows[k], c = cols[k];
  const float v = vals[k];
  const float* src = xt + (size_t)c * BATCH;
  float* dst = yt + (size_t)r * BATCH;
#pragma unroll
  for (int i = 0; i < BATCH; i += 32)
    atomicAdd(&dst[i + lane], v * src[i + lane]);
}

// ---------------- y = bc + y*flag (transposed layout) -----------------------
__global__ __launch_bounds__(256) void k_bcmask(float* __restrict__ yt,
                                                const float* __restrict__ bc,
                                                const float* __restrict__ flag) {
  const int n = blockIdx.x, b = threadIdx.x;
  const size_t i = (size_t)n * BATCH + b;
  yt[i] = bc[n] + yt[i] * flag[n];
}

// ---------------- staging helpers -------------------------------------------
__device__ __forceinline__ void stage_w(const float* __restrict__ W, int n0,
                                        int k0, float* lWf, int t) {
  const int row = t >> 3, q = t & 7;                    // 32 rows x 8 x float4
  const float* gp = W + (size_t)(n0 + row) * N + k0 + q * 4;
  float* lp = lWf + row * WROW + q * 4;
#if HAS_ASYNC_LDS
  __builtin_amdgcn_global_load_async_to_lds_b128((v4i*)gp, (v4i*)lp, 0, 0);
#else
  *(float4*)lp = *(const float4*)gp;
#endif
}

__device__ __forceinline__ void stage_h(const float* __restrict__ Hin, int k0,
                                        __bf16* hhi, __bf16* hlo, int t) {
  const float* src = Hin + (size_t)k0 * BATCH + t;      // fixed b = t
  __bf16* dh = hhi + t * LROW;
  __bf16* dl = hlo + t * LROW;
#pragma unroll
  for (int kk = 0; kk < KB; kk += 2) {
    const float f0 = src[(size_t)kk * BATCH];
    const float f1 = src[(size_t)(kk + 1) * BATCH];
    bf16x2 ph = pk_bf16(f0, f1);
    bf16x2 pl = pk_bf16(f0 - bf2f(ph[0]), f1 - bf2f(ph[1]));
    *(bf16x2*)(dh + kk) = ph;
    *(bf16x2*)(dl + kk) = pl;
  }
}

// ---------------- GEMM: Out_t[n,b] = sum_k W[n,k]*H_t[k,b] + bias[n] --------
// MODE 0: out [N][BATCH] with SELU.  MODE 1: out [BATCH][N] with bc+flag mask.
template <int MODE>
__global__ __launch_bounds__(256)
void k_gemm_bf16x3(const float* __restrict__ Hin,   // [N][BATCH]
                   const float* __restrict__ W,     // [N][N]
                   const float* __restrict__ bias,
                   const float* __restrict__ bc,
                   const float* __restrict__ flag,
                   float* __restrict__ out) {
  extern __shared__ __align__(16) char smem[];
  float*  lWf  = (float*)smem;                                  // [2][WBUF] fp32
  __bf16* lHhi = (__bf16*)(smem + 2 * WBUF * sizeof(float));    // [2][HBUF]
  __bf16* lHlo = lHhi + 2 * HBUF;                               // [2][HBUF]

  const int t    = threadIdx.x;
  const int wave = t >> 5;
  const int lane = t & 31;
  const int n0   = blockIdx.x * BM;
  const int n16  = wave >> 2;        // 16-row strip of W tile (0..1)
  const int btg  = wave & 3;         // group of 4 b-tiles
  const int half = lane >> 4;
  const int m    = lane & 15;

  v8f acc[4] = {};

  // prologue: stage k0 = 0 into buffer 0
  stage_w(W, n0, 0, lWf, t);
  stage_h(Hin, 0, lHhi, lHlo, t);
  async_wait();
  __syncthreads();

  for (int k0 = 0; k0 < N; k0 += KB) {
    const int cur = (k0 / KB) & 1;
    const int nxt = cur ^ 1;
    if (k0 + KB < N) {                       // pipeline next stage
      stage_w(W, n0, k0 + KB, lWf + nxt * WBUF, t);
      stage_h(Hin, k0 + KB, lHhi + nxt * HBUF, lHlo + nxt * HBUF, t);
    }

    // ---- A fragments: fp32 W tile -> bf16 hi/lo (16x32, 16-bit layout) ----
    // lane<16: K 0..7 & 16..23 of row m; lane>=16: K 8..15 & 24..31
    const float* wb = lWf + cur * WBUF + (n16 * 16 + m) * WROW + half * 8;
    FragBf Ahi, Alo;
    cvt_split4(*(const float4*)(wb),      Ahi.c[0], Alo.c[0]);
    cvt_split4(*(const float4*)(wb + 4),  Ahi.c[1], Alo.c[1]);
    cvt_split4(*(const float4*)(wb + 16), Ahi.c[2], Alo.c[2]);
    cvt_split4(*(const float4*)(wb + 20), Ahi.c[3], Alo.c[3]);

#pragma unroll
    for (int j = 0; j < 4; ++j) {
      // B fragment (32x16): lane<16 holds K 0..7 & 16..23 of column m; lane>=16 K 8..15 & 24..31
      const int b = (btg * 4 + j) * 16 + m;
      const __bf16* hr = lHhi + cur * HBUF + b * LROW + half * 8;
      const __bf16* hl = lHlo + cur * HBUF + b * LROW + half * 8;
      FragBf Bhi, Blo;
      Bhi.c[0] = ((const bf16x4*)hr)[0]; Bhi.c[1] = ((const bf16x4*)hr)[1];
      Bhi.c[2] = ((const bf16x4*)hr)[4]; Bhi.c[3] = ((const bf16x4*)hr)[5];
      Blo.c[0] = ((const bf16x4*)hl)[0]; Blo.c[1] = ((const bf16x4*)hl)[1];
      Blo.c[2] = ((const bf16x4*)hl)[4]; Blo.c[3] = ((const bf16x4*)hl)[5];
      acc[j] = __builtin_amdgcn_wmma_f32_16x16x32_bf16(
          false, Alo.v, false, Bhi.v, (short)0, acc[j], false, false);
      acc[j] = __builtin_amdgcn_wmma_f32_16x16x32_bf16(
          false, Ahi.v, false, Blo.v, (short)0, acc[j], false, false);
      acc[j] = __builtin_amdgcn_wmma_f32_16x16x32_bf16(
          false, Ahi.v, false, Bhi.v, (short)0, acc[j], false, false);
    }

    async_wait();          // this wave's async W copies for `nxt` done
    __syncthreads();       // everyone done computing `cur` + staging `nxt`
  }

  // ---- epilogue: lane L, VGPR r -> n = n0+n16*16+r+8*half, b-col = m ----
  float bs[8];
#pragma unroll
  for (int r = 0; r < 8; ++r) bs[r] = bias[n0 + n16 * 16 + r + 8 * half];
#pragma unroll
  for (int j = 0; j < 4; ++j) {
    const int b = (btg * 4 + j) * 16 + m;
#pragma unroll
    for (int r = 0; r < 8; ++r) {
      const int n = n0 + n16 * 16 + r + 8 * half;
      const float v = acc[j][r] + bs[r];
      if (MODE == 0) {
        out[(size_t)n * BATCH + b] = selu_f(v);         // stay transposed
      } else {
        out[(size_t)b * N + n] = bc[n] + flag[n] * v;   // un-transpose + mask
      }
    }
  }
}

// ---------------------------------------------------------------------------
extern "C" void kernel_launch(void* const* d_in, const int* in_sizes, int n_in,
                              void* d_out, int out_size, void* d_ws, size_t ws_size,
                              hipStream_t stream) {
  const float* x    = (const float*)d_in[0];
  const float* vals = (const float*)d_in[1];
  const float* bc   = (const float*)d_in[2];
  const float* flag = (const float*)d_in[3];
  const float* W1   = (const float*)d_in[4];
  const float* b1   = (const float*)d_in[5];
  const float* W2   = (const float*)d_in[6];
  const float* b2   = (const float*)d_in[7];
  const float* W3   = (const float*)d_in[8];
  const float* b3   = (const float*)d_in[9];
  const int*   rows = (const int*)d_in[10];
  const int*   cols = (const int*)d_in[11];

  float* buf0 = (float*)d_ws;                 // x_t, later h1   [N][B]
  float* buf1 = buf0 + (size_t)N * BATCH;     // y_t, later h2   [N][B]

  const size_t smem = 2 * WBUF * sizeof(float) + 4 * HBUF * sizeof(__bf16);

  k_transpose<<<dim3(N / 32, BATCH / 32), 256, 0, stream>>>(x, buf0);
  (void)hipMemsetAsync(buf1, 0, (size_t)N * BATCH * sizeof(float), stream);
  k_spmv<<<NNZ / 8, 256, 0, stream>>>(rows, cols, vals, buf0, buf1);
  k_bcmask<<<N, 256, 0, stream>>>(buf1, bc, flag);

  k_gemm_bf16x3<0><<<N / BM, 256, smem, stream>>>(buf1, W1, b1, nullptr, nullptr, buf0);
  k_gemm_bf16x3<0><<<N / BM, 256, smem, stream>>>(buf0, W2, b2, nullptr, nullptr, buf1);
  k_gemm_bf16x3<1><<<N / BM, 256, smem, stream>>>(buf1, W3, b3, bc, flag, (float*)d_out);
}
